// MSACore_16054587752604
// MI455X (gfx1250) — matrix operand order repaired and checked by reading, not compiled, not run on hardware
//
#include <hip/hip_runtime.h>
#include <math.h>

typedef __bf16 bf16_t;
typedef __attribute__((ext_vector_type(16))) __bf16 bf16x16;
typedef __attribute__((ext_vector_type(8)))  __bf16 bf16x8;
typedef __attribute__((ext_vector_type(8)))  float  floatx8;

#define S_DIM 128
#define R_DIM 256
#define DN    256
#define DP    128
#define NH    8
#define CH    32
#define HCD   256
#define DFFD  1024
#define NTOK  (S_DIM * R_DIM)

// ---------------------------------------------------------------------------
// WMMA helpers (this code's contiguous k order k = 16*half + s, same in A and B)
// ---------------------------------------------------------------------------
__device__ __forceinline__ floatx8 wmma_bf16f32(bf16x16 a, bf16x16 b, floatx8 c) {
  return __builtin_amdgcn_wmma_f32_16x16x32_bf16(false, a, false, b, (short)0, c,
                                                 false, false);
}

// A-matrix 16x32 bf16 fragment: lane half 0 holds K 0..15, half 1
// holds K 16..31 of row (lane&15). rowk points at row start + k0.
__device__ __forceinline__ bf16x16 load_a_frag(const bf16_t* rowk, int half) {
  return *(const bf16x16*)(rowk + half * 16);
}
// B-matrix 32x16 bf16 fragment: lanes 0-15 hold K 0..15 of col (lane&15),
// lanes 16-31 hold K 16..31. rowk points at B row (= output col) start + k0.
__device__ __forceinline__ bf16x16 load_b_frag(const bf16_t* rowk, int half) {
  return *(const bf16x16*)(rowk + half * 16);
}

// ---------------------------------------------------------------------------
// Generic WMMA GEMM: C[M,N] = A[M,K] * B[N,K]^T.
// 256 threads = 8 waves (2 x 4), each wave computes a 32x32 tile as 2x2
// 16x16 WMMA subtiles (4 WMMAs per 4 fragment loads per K-chunk).
// Block tile: 64 x 128. Epilogue selected by template.
// ---------------------------------------------------------------------------
constexpr int EPI_BF16 = 0;  // outB = acc*scale
constexpr int EPI_SIG  = 1;  // outB = sigmoid(acc + bias[n])
constexpr int EPI_RELU = 2;  // outB = relu(acc + bias[n])
constexpr int EPI_VT   = 3;  // outB[Vt[s][h][c][r]] = acc   (inner = tokens/group)
constexpr int EPI_RES  = 4;  // outF[perm(m),n] = resid + acc + bias[n]

template<int MODE>
__global__ __launch_bounds__(256)
void gemm_bf16_wmma(const bf16_t* __restrict__ A, const bf16_t* __restrict__ B,
                    int Mdim, int Ndim, int Kdim, float scale,
                    const float* __restrict__ bias,
                    const float* __restrict__ resid,
                    float* __restrict__ outF,
                    bf16_t* __restrict__ outB,
                    int inner, int outer)
{
  const int lane = threadIdx.x & 31;
  const int wave = threadIdx.x >> 5;
  const int m0 = blockIdx.x * 64 + (wave & 1) * 32;
  const int n0 = blockIdx.y * 128 + (wave >> 1) * 32;
  const int r = lane & 15, half = lane >> 4;
  const bf16_t* ap0 = A + (size_t)(m0 + r) * Kdim;
  const bf16_t* ap1 = A + (size_t)(m0 + 16 + r) * Kdim;
  const bf16_t* bp0 = B + (size_t)(n0 + r) * Kdim;
  const bf16_t* bp1 = B + (size_t)(n0 + 16 + r) * Kdim;
  floatx8 acc[2][2] = {};
  for (int k = 0; k < Kdim; k += 32) {
    // CDNA5 speculative prefetch (global_prefetch_b8); OOB is silently dropped.
    __builtin_prefetch(ap0 + k + 128, 0, 1);
    __builtin_prefetch(bp0 + k + 128, 0, 1);
    const bf16x16 a0 = load_a_frag(ap0 + k, half);
    const bf16x16 a1 = load_a_frag(ap1 + k, half);
    const bf16x16 b0 = load_b_frag(bp0 + k, half);
    const bf16x16 b1 = load_b_frag(bp1 + k, half);
    acc[0][0] = wmma_bf16f32(a0, b0, acc[0][0]);
    acc[0][1] = wmma_bf16f32(a0, b1, acc[0][1]);
    acc[1][0] = wmma_bf16f32(a1, b0, acc[1][0]);
    acc[1][1] = wmma_bf16f32(a1, b1, acc[1][1]);
  }
  // D layout: VGPR j -> M = j + 8*half (within 16x16 tile), N = lane&15
#pragma unroll
  for (int sm = 0; sm < 2; ++sm) {
#pragma unroll
    for (int sn = 0; sn < 2; ++sn) {
#pragma unroll
      for (int j = 0; j < 8; ++j) {
        const int m = m0 + sm * 16 + j + half * 8;
        const int n = n0 + sn * 16 + r;
        float v = acc[sm][sn][j] * scale;
        if (MODE == EPI_BF16) {
          outB[(size_t)m * Ndim + n] = (bf16_t)v;
        } else if (MODE == EPI_SIG) {
          v += bias[n];
          outB[(size_t)m * Ndim + n] = (bf16_t)(1.0f / (1.0f + __expf(-v)));
        } else if (MODE == EPI_RELU) {
          v += bias[n];
          outB[(size_t)m * Ndim + n] = (bf16_t)fmaxf(v, 0.0f);
        } else if (MODE == EPI_VT) {
          const int grp = m / inner, rr = m - grp * inner;
          const int h = n >> 5, c = n & 31;
          outB[((size_t)((grp * NH + h) * CH + c)) * inner + rr] = (bf16_t)v;
        } else { // EPI_RES
          size_t row;
          if (inner > 0) { const int g2 = m / inner, rr = m - g2 * inner;
                           row = (size_t)rr * outer + g2; }
          else           { row = (size_t)m; }
          const size_t idx = row * Ndim + n;
          outF[idx] = resid[idx] + v + bias[n];
        }
      }
    }
  }
}

// ---------------------------------------------------------------------------
// LayerNorm over last dim (blockDim == dim, power of 2), fp32 -> bf16.
// Optional transposed output row mapping: orow = (row%inner)*outer + row/inner.
// ---------------------------------------------------------------------------
__global__ void ln_bf16_kernel(const float* __restrict__ x,
                               const float* __restrict__ g,
                               const float* __restrict__ b,
                               bf16_t* __restrict__ out,
                               int dim, int inner, int outer)
{
  __shared__ float red[256];
  const int row = blockIdx.x;
  const int t = threadIdx.x;
  const float v = x[(size_t)row * dim + t];
  red[t] = v;
  __syncthreads();
  for (int s = blockDim.x >> 1; s > 0; s >>= 1) {
    if (t < s) red[t] += red[t + s];
    __syncthreads();
  }
  const float mean = red[0] / (float)dim;
  __syncthreads();
  const float dv = v - mean;
  red[t] = dv * dv;
  __syncthreads();
  for (int s = blockDim.x >> 1; s > 0; s >>= 1) {
    if (t < s) red[t] += red[t + s];
    __syncthreads();
  }
  const float var = red[0] / (float)dim;
  const float y = dv * rsqrtf(var + 1e-5f) * g[t] + b[t];
  int orow = row;
  if (inner > 0) orow = (row % inner) * outer + (row / inner);
  out[(size_t)orow * dim + t] = (bf16_t)y;
}

// ---------------------------------------------------------------------------
// Pair bias: per (q,k) row of pair[R,R,DP]: LN over DP, dot with row_b_w[H,DP]
// -> bias[h, q, k]. blockDim == DP (128).
// ---------------------------------------------------------------------------
__global__ void pair_bias_kernel(const float* __restrict__ Z,
                                 const float* __restrict__ g,
                                 const float* __restrict__ b,
                                 const float* __restrict__ bw,
                                 float* __restrict__ bias, int RR)
{
  __shared__ float red[DP];
  __shared__ float zn[DP];
  const int row = blockIdx.x;
  const int t = threadIdx.x;
  const float v = Z[(size_t)row * DP + t];
  red[t] = v;
  __syncthreads();
  for (int s = DP >> 1; s > 0; s >>= 1) {
    if (t < s) red[t] += red[t + s];
    __syncthreads();
  }
  const float mean = red[0] / (float)DP;
  __syncthreads();
  const float dv = v - mean;
  red[t] = dv * dv;
  __syncthreads();
  for (int s = DP >> 1; s > 0; s >>= 1) {
    if (t < s) red[t] += red[t + s];
    __syncthreads();
  }
  const float var = red[0] / (float)DP;
  __syncthreads();
  zn[t] = dv * rsqrtf(var + 1e-5f) * g[t] + b[t];
  __syncthreads();
  for (int h = 0; h < NH; ++h) {
    red[t] = zn[t] * bw[h * DP + t];
    __syncthreads();
    for (int s = DP >> 1; s > 0; s >>= 1) {
      if (t < s) red[t] += red[t + s];
      __syncthreads();
    }
    if (t == 0) bias[(size_t)h * RR + row] = red[0];
    __syncthreads();
  }
}

// ---------------------------------------------------------------------------
// Fused attention: one block (4 waves / 128 thr) = one (group, head, 16-query
// tile). Logits via WMMA (+bias+mask) into LDS, parallel softmax, P*Vt^T via
// WMMA with cross-wave reduce, gate, bf16 store. NQ == NK tokens per group.
// ---------------------------------------------------------------------------
template<bool HAS_BIAS>
__global__ __launch_bounds__(128)
void attn_fused_kernel(const bf16_t* __restrict__ Q, const bf16_t* __restrict__ K,
                       const bf16_t* __restrict__ Vt, const bf16_t* __restrict__ Gate,
                       bf16_t* __restrict__ O,
                       const float* __restrict__ bias,   // [H,NQ,NK] or null
                       const float* __restrict__ mask,
                       int NQ, int NK, int mask_gs, int mask_es)
{
  __shared__ __align__(32) float  slog[16 * 256];
  __shared__ __align__(32) bf16_t sp[16 * 256];
  __shared__ __align__(32) float  sred[2 * 16 * 16]; // also softmax partials

  const int qtiles = NQ >> 4;
  const int qt  = blockIdx.x % qtiles;
  const int h   = (blockIdx.x / qtiles) % NH;
  const int grp = blockIdx.x / (qtiles * NH);
  const int lane = threadIdx.x & 31;
  const int wave = threadIdx.x >> 5;
  const int r = lane & 15, half = lane >> 4;

  // Q fragment (A operand, K-dim = CH = 32): row = query (lane&15)
  const bf16_t* qrow = Q + (size_t)(grp * NQ + qt * 16 + r) * HCD + h * CH;
  const bf16x16 qf = load_a_frag(qrow, half);

  const int ktiles = NK >> 4;
  for (int kt = wave; kt < ktiles; kt += 4) {
    const bf16_t* krow = K + (size_t)(grp * NK + kt * 16 + r) * HCD + h * CH;
    const bf16x16 kf = load_b_frag(krow, half);
    floatx8 acc = {};
    acc = wmma_bf16f32(qf, kf, acc);
    const int kk = kt * 16 + r;                 // key column of this lane
    const float mterm =
        1e9f * (mask[(size_t)grp * mask_gs + (size_t)kk * mask_es] - 1.0f);
    const float* brow = HAS_BIAS
        ? bias + ((size_t)h * NQ + qt * 16) * NK + kk
        : nullptr;
#pragma unroll
    for (int j = 0; j < 8; ++j) {
      const int m = j + half * 8;               // query within tile
      float lv = acc[j] + mterm;
      if (HAS_BIAS) lv += brow[(size_t)m * NK];
      slog[m * 256 + kk] = lv;
    }
  }
  __syncthreads();

  // Parallel softmax: 8 threads per row (128 threads / 16 rows).
  {
    const int mrow = threadIdx.x >> 3;
    const int sub = threadIdx.x & 7;
    float mx = -1e30f;
    for (int k = sub; k < NK; k += 8) mx = fmaxf(mx, slog[mrow * 256 + k]);
    sred[mrow * 8 + sub] = mx;
    __syncthreads();
    for (int s = 4; s > 0; s >>= 1) {
      if (sub < s)
        sred[mrow * 8 + sub] = fmaxf(sred[mrow * 8 + sub], sred[mrow * 8 + sub + s]);
      __syncthreads();
    }
    mx = sred[mrow * 8];
    __syncthreads();
    float sum = 0.0f;
    for (int k = sub; k < NK; k += 8) {
      const float e = __expf(slog[mrow * 256 + k] - mx);
      slog[mrow * 256 + k] = e;
      sum += e;
    }
    sred[mrow * 8 + sub] = sum;
    __syncthreads();
    for (int s = 4; s > 0; s >>= 1) {
      if (sub < s) sred[mrow * 8 + sub] += sred[mrow * 8 + sub + s];
      __syncthreads();
    }
    const float inv = 1.0f / sred[mrow * 8];
    for (int k = sub; k < NK; k += 8)
      sp[mrow * 256 + k] = (bf16_t)(slog[mrow * 256 + k] * inv);
  }
  __syncthreads();

  // O[16, CH] = P[16, NK] * V; Vt layout [grp][h][c][token]. ct = c-tile.
  const int ct = wave & 1;
  floatx8 oacc = {};
  const int nchunks = NK >> 5;
  for (int kc = (wave >> 1); kc < nchunks; kc += 2) {
    const bf16x16 pf = load_a_frag(&sp[r * 256 + kc * 32], half);
    const bf16_t* vrow =
        Vt + (size_t)((grp * NH + h) * CH + ct * 16 + r) * NK + kc * 32;
    const bf16x16 vf = load_b_frag(vrow, half);
    oacc = wmma_bf16f32(pf, vf, oacc);
  }
  if (wave >= 2) {
#pragma unroll
    for (int j = 0; j < 8; ++j)
      sred[(ct * 16 + j + half * 8) * 16 + r] = oacc[j];
  }
  __syncthreads();
  if (wave < 2) {
#pragma unroll
    for (int j = 0; j < 8; ++j) {
      const int m = j + half * 8;
      const int n = ct * 16 + r;
      const float v = oacc[j] + sred[(ct * 16 + m) * 16 + r];
      const size_t gidx = (size_t)(grp * NQ + qt * 16 + m) * HCD + h * CH + n;
      const float gv = (float)Gate[gidx];
      O[gidx] = (bf16_t)(v * gv);
    }
  }
}

// ---------------------------------------------------------------------------
__global__ void f32_to_bf16_kernel(const float* __restrict__ in,
                                   bf16_t* __restrict__ out, size_t n)
{
  const size_t i = (size_t)blockIdx.x * blockDim.x + threadIdx.x;
  if (i < n) out[i] = (bf16_t)in[i];
}

// ---------------------------------------------------------------------------
extern "C" void kernel_launch(void* const* d_in, const int* in_sizes, int n_in,
                              void* d_out, int out_size, void* d_ws, size_t ws_size,
                              hipStream_t stream)
{
  const float* node        = (const float*)d_in[0];
  const float* pair        = (const float*)d_in[1];
  const float* node_mask   = (const float*)d_in[2];
  const float* row_ln_m_g  = (const float*)d_in[3];
  const float* row_ln_m_b  = (const float*)d_in[4];
  const float* row_ln_z_g  = (const float*)d_in[5];
  const float* row_ln_z_b  = (const float*)d_in[6];
  const float* row_b_w     = (const float*)d_in[7];
  const float* row_wq      = (const float*)d_in[8];
  const float* row_wk      = (const float*)d_in[9];
  const float* row_wv      = (const float*)d_in[10];
  const float* row_wg      = (const float*)d_in[11];
  const float* row_bg      = (const float*)d_in[12];
  const float* row_wo      = (const float*)d_in[13];
  const float* row_out_bias= (const float*)d_in[14];
  const float* col_ln_g    = (const float*)d_in[15];
  const float* col_ln_b    = (const float*)d_in[16];
  const float* col_wq      = (const float*)d_in[17];
  const float* col_wk      = (const float*)d_in[18];
  const float* col_wv      = (const float*)d_in[19];
  const float* col_wg      = (const float*)d_in[20];
  const float* col_bg      = (const float*)d_in[21];
  const float* col_wo      = (const float*)d_in[22];
  const float* col_bo      = (const float*)d_in[23];
  const float* tr_ln_g     = (const float*)d_in[24];
  const float* tr_ln_b     = (const float*)d_in[25];
  const float* tr_w1       = (const float*)d_in[26];
  const float* tr_b1       = (const float*)d_in[27];
  const float* tr_w2       = (const float*)d_in[28];
  const float* tr_b2       = (const float*)d_in[29];
  float* out = (float*)d_out;

  // ---- workspace carve-up ----
  char* ws = (char*)d_ws;
  size_t off = 0;
  auto alloc = [&](size_t bytes) -> char* {
    char* p = ws + off;
    off += (bytes + 255) & ~(size_t)255;
    return p;
  };
  const size_t WSML = (size_t)HCD * DN * sizeof(bf16_t);       // 128 KB
  bf16_t* wq_r = (bf16_t*)alloc(WSML);
  bf16_t* wk_r = (bf16_t*)alloc(WSML);
  bf16_t* wv_r = (bf16_t*)alloc(WSML);
  bf16_t* wg_r = (bf16_t*)alloc(WSML);
  bf16_t* wo_r = (bf16_t*)alloc(WSML);
  bf16_t* wq_c = (bf16_t*)alloc(WSML);
  bf16_t* wk_c = (bf16_t*)alloc(WSML);
  bf16_t* wv_c = (bf16_t*)alloc(WSML);
  bf16_t* wg_c = (bf16_t*)alloc(WSML);
  bf16_t* wo_c = (bf16_t*)alloc(WSML);
  bf16_t* w1_b = (bf16_t*)alloc((size_t)DFFD * DN * sizeof(bf16_t));
  bf16_t* w2_b = (bf16_t*)alloc((size_t)DN * DFFD * sizeof(bf16_t));
  const size_t TOKD = (size_t)NTOK * HCD * sizeof(bf16_t);     // 16 MB
  bf16_t* Mb  = (bf16_t*)alloc(TOKD);
  bf16_t* Qb  = (bf16_t*)alloc(TOKD);
  bf16_t* Kb  = (bf16_t*)alloc(TOKD);
  bf16_t* Vtb = (bf16_t*)alloc(TOKD);
  bf16_t* Gb  = (bf16_t*)alloc(TOKD);
  bf16_t* Ob  = (bf16_t*)alloc(TOKD);
  float*  biasB = (float*)alloc((size_t)NH * R_DIM * R_DIM * sizeof(float)); // 2 MB
  bf16_t* Hb  = (bf16_t*)alloc((size_t)NTOK * DFFD * sizeof(bf16_t));        // 64 MB
  (void)ws_size; (void)n_in; (void)in_sizes; (void)out_size;

  auto cvt = [&](const float* src, bf16_t* dst, size_t n) {
    f32_to_bf16_kernel<<<(unsigned)((n + 255) / 256), 256, 0, stream>>>(src, dst, n);
  };
  cvt(row_wq, wq_r, (size_t)HCD * DN);
  cvt(row_wk, wk_r, (size_t)HCD * DN);
  cvt(row_wv, wv_r, (size_t)HCD * DN);
  cvt(row_wg, wg_r, (size_t)HCD * DN);
  cvt(row_wo, wo_r, (size_t)DN * HCD);
  cvt(col_wq, wq_c, (size_t)HCD * DN);
  cvt(col_wk, wk_c, (size_t)HCD * DN);
  cvt(col_wv, wv_c, (size_t)HCD * DN);
  cvt(col_wg, wg_c, (size_t)HCD * DN);
  cvt(col_wo, wo_c, (size_t)DN * HCD);
  cvt(tr_w1, w1_b, (size_t)DFFD * DN);
  cvt(tr_w2, w2_b, (size_t)DN * DFFD);

  const float qscale = 0.17677669529663687f; // 1/sqrt(32)
  const dim3 gHC(NTOK / 64, HCD / 128);
  const dim3 gD (NTOK / 64, DN / 128);
  const dim3 gFF(NTOK / 64, DFFD / 128);

  // ================= Row attention (pair-biased) =================
  ln_bf16_kernel<<<NTOK, DN, 0, stream>>>(node, row_ln_m_g, row_ln_m_b, Mb, DN, 0, 0);
  pair_bias_kernel<<<R_DIM * R_DIM, DP, 0, stream>>>(pair, row_ln_z_g, row_ln_z_b,
                                                     row_b_w, biasB, R_DIM * R_DIM);
  gemm_bf16_wmma<EPI_BF16><<<gHC, 256, 0, stream>>>(Mb, wq_r, NTOK, HCD, DN, qscale,
      nullptr, nullptr, nullptr, Qb, 0, 0);
  gemm_bf16_wmma<EPI_BF16><<<gHC, 256, 0, stream>>>(Mb, wk_r, NTOK, HCD, DN, 1.0f,
      nullptr, nullptr, nullptr, Kb, 0, 0);
  gemm_bf16_wmma<EPI_VT><<<gHC, 256, 0, stream>>>(Mb, wv_r, NTOK, HCD, DN, 1.0f,
      nullptr, nullptr, nullptr, Vtb, R_DIM, 0);
  gemm_bf16_wmma<EPI_SIG><<<gHC, 256, 0, stream>>>(Mb, wg_r, NTOK, HCD, DN, 1.0f,
      row_bg, nullptr, nullptr, Gb, 0, 0);
  attn_fused_kernel<true><<<S_DIM * NH * (R_DIM / 16), 128, 0, stream>>>(
      Qb, Kb, Vtb, Gb, Ob, biasB, node_mask,
      R_DIM, R_DIM, /*mask_gs=*/R_DIM, /*mask_es=*/1);
  gemm_bf16_wmma<EPI_RES><<<gD, 256, 0, stream>>>(Ob, wo_r, NTOK, DN, HCD, 1.0f,
      row_out_bias, node, out, nullptr, 0, 0);

  // ================= Column attention =================
  // LN + transpose [S,R,d] -> [R,S,d]
  ln_bf16_kernel<<<NTOK, DN, 0, stream>>>(out, col_ln_g, col_ln_b, Mb, DN,
                                          /*inner=*/R_DIM, /*outer=*/S_DIM);
  gemm_bf16_wmma<EPI_BF16><<<gHC, 256, 0, stream>>>(Mb, wq_c, NTOK, HCD, DN, qscale,
      nullptr, nullptr, nullptr, Qb, 0, 0);
  gemm_bf16_wmma<EPI_BF16><<<gHC, 256, 0, stream>>>(Mb, wk_c, NTOK, HCD, DN, 1.0f,
      nullptr, nullptr, nullptr, Kb, 0, 0);
  gemm_bf16_wmma<EPI_VT><<<gHC, 256, 0, stream>>>(Mb, wv_c, NTOK, HCD, DN, 1.0f,
      nullptr, nullptr, nullptr, Vtb, S_DIM, 0);
  gemm_bf16_wmma<EPI_SIG><<<gHC, 256, 0, stream>>>(Mb, wg_c, NTOK, HCD, DN, 1.0f,
      col_bg, nullptr, nullptr, Gb, 0, 0);
  attn_fused_kernel<false><<<R_DIM * NH * (S_DIM / 16), 128, 0, stream>>>(
      Qb, Kb, Vtb, Gb, Ob, nullptr, node_mask,
      S_DIM, S_DIM, /*mask_gs=*/1, /*mask_es=*/R_DIM);
  // output projection with [R,S] -> [S,R] permuted residual store
  gemm_bf16_wmma<EPI_RES><<<gD, 256, 0, stream>>>(Ob, wo_c, NTOK, DN, HCD, 1.0f,
      col_bo, out, out, nullptr, /*inner=*/S_DIM, /*outer=*/R_DIM);

  // ================= Transition =================
  ln_bf16_kernel<<<NTOK, DN, 0, stream>>>(out, tr_ln_g, tr_ln_b, Mb, DN, 0, 0);
  gemm_bf16_wmma<EPI_RELU><<<gFF, 256, 0, stream>>>(Mb, w1_b, NTOK, DFFD, DN, 1.0f,
      tr_b1, nullptr, nullptr, Hb, 0, 0);
  gemm_bf16_wmma<EPI_RES><<<gD, 256, 0, stream>>>(Hb, w2_b, NTOK, DN, DFFD, 1.0f,
      tr_b2, out, out, nullptr, 0, 0);
}
